// RuntimeExaone4Attention_59347858096267
// MI455X (gfx1250) — 16-bit and 8-bit matrix operand order repaired and checked by reading, compiled, not run on hardware
//
#include <hip/hip_runtime.h>

typedef int      v8i  __attribute__((ext_vector_type(8)));
typedef float    v8f  __attribute__((ext_vector_type(8)));
typedef _Float16 v16h __attribute__((ext_vector_type(16)));

#define S_LEN 1024
#define NHQ   32
#define NKVH  8
#define HDm   64

#define OFF_PW_QKV   ((size_t)0)
#define OFF_PW_O     ((size_t)6291456)
#define OFF_SCL_QKV  ((size_t)10485760)
#define OFF_SCL_O    ((size_t)10682368)
#define OFF_WSUM_QKV ((size_t)10813440)
#define OFF_WSUM_O   ((size_t)10825728)
#define OFF_XQ       ((size_t)10833920)
#define OFF_XS       ((size_t)12931072)
#define OFF_XZ       ((size_t)12935168)
#define OFF_QKVF     ((size_t)12939264)
#define OFF_QQ       ((size_t)25522176)
#define OFF_QS       ((size_t)27619328)
#define OFF_QZ       ((size_t)27750400)
#define OFF_QSM      ((size_t)27881472)
#define OFF_KQ       ((size_t)28012544)
#define OFF_KS       ((size_t)28536832)
#define OFF_KZ       ((size_t)28569600)
#define OFF_KSM      ((size_t)28602368)
#define OFF_VT       ((size_t)28635136)
#define OFF_AO       ((size_t)29683712)
#define OFF_AQ       ((size_t)38072320)
#define OFF_AS       ((size_t)40169472)
#define OFF_AZ       ((size_t)40173568)

__global__ void __launch_bounds__(128) pack_weights_kernel(
    const int* q_qw, const float* q_sc, const int* q_zp,
    const int* k_qw, const float* k_sc, const int* k_zp,
    const int* v_qw, const float* v_sc, const int* v_zp,
    const int* o_qw, const float* o_sc, const int* o_zp,
    signed char* pw_qkv, float* scl_qkv, float* wsum_qkv,
    signed char* pw_o,   float* scl_o,   float* wsum_o)
{
  __shared__ float red[128];
  int n = blockIdx.x;
  const int* qw; const float* sc; const int* zp;
  signed char* pw; float* scl; float* ws;
  if (n < 3072) {
    if (n < 2048)      {            qw = q_qw + (size_t)n*2048; sc = q_sc + n*16; zp = q_zp + n*16; }
    else if (n < 2560) { int m=n-2048; qw = k_qw + (size_t)m*2048; sc = k_sc + m*16; zp = k_zp + m*16; }
    else               { int m=n-2560; qw = v_qw + (size_t)m*2048; sc = v_sc + m*16; zp = v_zp + m*16; }
    pw = pw_qkv + (size_t)n*2048; scl = scl_qkv + n*16; ws = wsum_qkv + n;
  } else {
    int m = n - 3072;
    qw = o_qw + (size_t)m*2048; sc = o_sc + m*16; zp = o_zp + m*16;
    pw = pw_o + (size_t)m*2048; scl = scl_o + m*16; ws = wsum_o + m;
  }
  int t = threadIdx.x;
  float part = 0.f;
  for (int idx = t; idx < 2048; idx += 128) {
    int g = idx >> 7;
    int c = qw[idx] - zp[g];
    pw[idx] = (signed char)c;
    part += sc[g] * (float)c;
  }
  red[t] = part; __syncthreads();
  for (int o = 64; o > 0; o >>= 1) { if (t < o) red[t] += red[t+o]; __syncthreads(); }
  if (t < 16) scl[t] = sc[t];
  if (t == 0) *ws = red[0];
}

__global__ void __launch_bounds__(256) quant_act_kernel(
    const float* __restrict__ src, unsigned char* __restrict__ q,
    float* __restrict__ s_out, float* __restrict__ z_out)
{
  __shared__ float smin[256], smax[256];
  int m = blockIdx.x, t = threadIdx.x;
  const float* x = src + (size_t)m*2048;
  float mn = 3.0e38f, mx = -3.0e38f;
  for (int i = t; i < 2048; i += 256) { float v = x[i]; mn = fminf(mn,v); mx = fmaxf(mx,v); }
  smin[t]=mn; smax[t]=mx; __syncthreads();
  for (int o=128;o>0;o>>=1){ if(t<o){smin[t]=fminf(smin[t],smin[t+o]); smax[t]=fmaxf(smax[t],smax[t+o]);} __syncthreads(); }
  float scale = fmaxf((smax[0]-smin[0]) * (1.0f/255.0f), 1e-5f);
  float zp = fminf(fmaxf(rintf(-smin[0]/scale), 0.f), 255.f);
  if (t==0){ s_out[m]=scale; z_out[m]=zp; }
  for (int i=t;i<2048;i+=256){
    float v = fminf(fmaxf(rintf(x[i]/scale)+zp, 0.f), 255.f);
    q[(size_t)m*2048 + i] = (unsigned char)v;
  }
}

__global__ void __launch_bounds__(256) gemm_u8s8_kernel(
    const unsigned char* __restrict__ xq, const float* __restrict__ xs, const float* __restrict__ xz,
    const signed char* __restrict__ pw, const float* __restrict__ scl, const float* __restrict__ wsum,
    float* __restrict__ out, int M, int N)
{
  int wid = blockIdx.x * 8 + (threadIdx.x >> 5);
  int ntiles = N >> 6;
  int mt = wid / ntiles, nt = wid % ntiles;
  if (mt >= (M >> 5)) return;
  int lane = threadIdx.x & 31;
  int half = lane >> 4, ln = lane & 15;
  int row0 = mt*32 + ln;
  int col0 = nt*64 + ln;
  const unsigned char* ax0 = xq + (size_t)row0 * 2048 + half*16;
  const unsigned char* ax1 = ax0 + (size_t)16 * 2048;
  const signed char* bw0 = pw + (size_t)col0 * 2048 + half*16;
  constexpr int aoff[8] = {0,4,8,12,32,36,40,44};
  constexpr int boff[8] = {0,4,8,12,32,36,40,44};
  v8f facc[8];
#pragma unroll
  for (int x = 0; x < 8; ++x) facc[x] = (v8f){0.f,0.f,0.f,0.f,0.f,0.f,0.f,0.f};

  for (int g = 0; g < 16; ++g) {
    v8i acc[8];
#pragma unroll
    for (int x = 0; x < 8; ++x) acc[x] = (v8i){0,0,0,0,0,0,0,0};
#pragma unroll
    for (int kk = 0; kk < 2; ++kk) {
      int kb = g*128 + kk*64;
      v8i A0, A1, Bm[4];
#pragma unroll
      for (int r = 0; r < 8; ++r) {
        A0[r] = *(const int*)(ax0 + kb + aoff[r]);
        A1[r] = *(const int*)(ax1 + kb + aoff[r]);
      }
#pragma unroll
      for (int c = 0; c < 4; ++c)
#pragma unroll
        for (int r = 0; r < 8; ++r)
          Bm[c][r] = *(const int*)(bw0 + (size_t)c*16*2048 + kb + boff[r]);
      __builtin_prefetch(ax0 + kb + 128, 0, 3);
      __builtin_prefetch(ax1 + kb + 128, 0, 3);
#pragma unroll
      for (int c = 0; c < 4; ++c)
        __builtin_prefetch(bw0 + (size_t)c*16*2048 + kb + 128, 0, 3);
#pragma unroll
      for (int c = 0; c < 4; ++c) {
        acc[c]   = __builtin_amdgcn_wmma_i32_16x16x64_iu8(false, A0, true, Bm[c], acc[c],   false, false);
        acc[4+c] = __builtin_amdgcn_wmma_i32_16x16x64_iu8(false, A1, true, Bm[c], acc[4+c], false, false);
      }
    }
#pragma unroll
    for (int c = 0; c < 4; ++c) {
      float sg = scl[(col0 + c*16)*16 + g];
#pragma unroll
      for (int r = 0; r < 8; ++r) {
        facc[c][r]   += sg * (float)acc[c][r];
        facc[4+c][r] += sg * (float)acc[4+c][r];
      }
    }
  }
#pragma unroll
  for (int c = 0; c < 4; ++c) {
    float wn = wsum[col0 + c*16];
#pragma unroll
    for (int mi = 0; mi < 2; ++mi)
#pragma unroll
      for (int r = 0; r < 8; ++r) {
        int rr = mt*32 + mi*16 + r + half*8;
        out[(size_t)rr*N + col0 + c*16] = xs[rr] * (facc[mi*4+c][r] - xz[rr]*wn);
      }
  }
}

__global__ void __launch_bounds__(256) rope_quant_kernel(
    const float* __restrict__ qkvf, const float* __restrict__ cosb, const float* __restrict__ sinb,
    unsigned char* __restrict__ qq, float* __restrict__ qs, float* __restrict__ qz, float* __restrict__ qsm,
    unsigned char* __restrict__ kq, float* __restrict__ ks, float* __restrict__ kz, float* __restrict__ ksm,
    _Float16* __restrict__ vT)
{
  int wid = blockIdx.x*8 + (threadIdx.x >> 5);
  int lane = threadIdx.x & 31;
  int d0 = lane, d1 = lane + 32;
  int type, h, s;
  if (wid < 32768)      { type = 0; h = wid >> 10;           s = wid & 1023; }
  else if (wid < 40960) { type = 1; h = (wid - 32768) >> 10; s = wid & 1023; }
  else                  { type = 2; h = (wid - 40960) >> 10; s = wid & 1023; }
  int cb = (type==0) ? h*64 : (type==1 ? 2048 + h*64 : 2560 + h*64);
  const float* xr = qkvf + (size_t)s*3072 + cb;
  float x0 = xr[d0], x1 = xr[d1];
  float r0, r1;
  if (type < 2) {
    float c0 = cosb[s*64+d0], c1 = cosb[s*64+d1];
    float s0 = sinb[s*64+d0], s1 = sinb[s*64+d1];
    r0 = x0*c0 - x1*s0;
    r1 = x1*c1 + x0*s1;
  } else { r0 = x0; r1 = x1; }
  float mn = fminf(r0,r1), mx = fmaxf(r0,r1);
  for (int o=16;o>0;o>>=1){ mn = fminf(mn,__shfl_xor(mn,o,32)); mx = fmaxf(mx,__shfl_xor(mx,o,32)); }
  float scale = fmaxf((mx-mn)*(1.0f/255.0f), 1e-5f);
  float zp = fminf(fmaxf(rintf(-mn/scale), 0.f), 255.f);
  float q0 = fminf(fmaxf(rintf(r0/scale)+zp, 0.f), 255.f);
  float q1 = fminf(fmaxf(rintf(r1/scale)+zp, 0.f), 255.f);
  size_t rowi = (size_t)h*1024 + s;
  if (type == 0) {
    qq[rowi*64+d0] = (unsigned char)q0; qq[rowi*64+d1] = (unsigned char)q1;
    float sm = q0 + q1;
    for (int o=16;o>0;o>>=1) sm += __shfl_xor(sm,o,32);
    if (lane==0){ qs[rowi]=scale; qz[rowi]=zp; qsm[rowi]=sm; }
  } else if (type == 1) {
    kq[rowi*64+d0] = (unsigned char)q0; kq[rowi*64+d1] = (unsigned char)q1;
    float sm = q0 + q1;
    for (int o=16;o>0;o>>=1) sm += __shfl_xor(sm,o,32);
    if (lane==0){ ks[rowi]=scale; kz[rowi]=zp; ksm[rowi]=sm; }
  } else {
    vT[((size_t)h*64 + d0)*1024 + s] = (_Float16)((q0 - zp)*scale);
    vT[((size_t)h*64 + d1)*1024 + s] = (_Float16)((q1 - zp)*scale);
  }
}

__global__ void __launch_bounds__(256) attn_kernel(
    const unsigned char* __restrict__ qq, const float* __restrict__ qs, const float* __restrict__ qz, const float* __restrict__ qsm,
    const unsigned char* __restrict__ kq, const float* __restrict__ ks, const float* __restrict__ kz, const float* __restrict__ ksm,
    const _Float16* __restrict__ vT, float* __restrict__ ao)
{
  int wid = blockIdx.x*8 + (threadIdx.x >> 5);
  int h = wid >> 6, it = wid & 63;
  int ib = it*16;
  int hkv = h >> 2;
  int lane = threadIdx.x & 31;
  int half = lane >> 4, ln = lane & 15;
  int i = ib + ln;
  size_t qrow = (size_t)h*1024 + i;
  float s1 = qs[qrow], z1 = qz[qrow], sm1 = qsm[qrow];

  constexpr int aoff[8] = {0,4,8,12,32,36,40,44};
  constexpr int boff[8] = {0,4,8,12,32,36,40,44};
  int ab = half*16;
  v8i Bq;
  {
    const unsigned char* qp = qq + qrow*64 + half*16;
#pragma unroll
    for (int r = 0; r < 8; ++r) Bq[r] = *(const int*)(qp + boff[r]);
  }
  const unsigned char* kbasep = kq + (size_t)hkv*1024*64;
  const float* ksb  = ks  + hkv*1024;
  const float* kzb  = kz  + hkv*1024;
  const float* ksmb = ksm + hkv*1024;
  const float SC = 0.125f;

  auto qk_tile = [&](int jb, float* xv) {
    v8i A;
    const unsigned char* kp = kbasep + (size_t)(jb + ln)*64 + ab;
#pragma unroll
    for (int r = 0; r < 8; ++r) A[r] = *(const int*)(kp + aoff[r]);
    v8i acc = {0,0,0,0,0,0,0,0};
    acc = __builtin_amdgcn_wmma_i32_16x16x64_iu8(false, A, false, Bq, acc, false, false);
#pragma unroll
    for (int r = 0; r < 8; ++r) {
      int j = jb + r + half*8;
      float z2 = kzb[j];
      float x = SC * s1 * ksb[j] * ((float)acc[r] - z1*ksmb[j] - z2*sm1 + 64.f*z1*z2);
      if (j > i) x = -1.0e30f;
      xv[r] = x;
    }
  };

  float m = -3.0e38f, l = 0.f;
  for (int jb = 0; jb <= ib; jb += 16) {
    float xv[8];
    qk_tile(jb, xv);
    float tm = -3.0e38f;
#pragma unroll
    for (int r = 0; r < 8; ++r) tm = fmaxf(tm, xv[r]);
    float mnew = fmaxf(m, tm), lsum = 0.f;
#pragma unroll
    for (int r = 0; r < 8; ++r) lsum += __expf(xv[r] - mnew);
    l = l*__expf(m - mnew) + lsum;
    m = mnew;
  }
  {
    float om = __shfl_xor(m, 16, 32), ol = __shfl_xor(l, 16, 32);
    float m2 = fmaxf(m, om);
    l = l*__expf(m - m2) + ol*__expf(om - m2);
    m = m2;
  }
  float linv = 1.0f / l;

  v8f oacc[4];
#pragma unroll
  for (int dt = 0; dt < 4; ++dt) oacc[dt] = (v8f){0.f,0.f,0.f,0.f,0.f,0.f,0.f,0.f};
  constexpr int joff0[8] = {0,2,4,6,8,10,12,14};
  constexpr int joff1[8] = {16,18,20,22,24,26,28,30};
  for (int jb = 0; jb <= ib; jb += 32) {
    float pA[8], pB[8], xv[8];
    qk_tile(jb, xv);
#pragma unroll
    for (int r = 0; r < 8; ++r)
      pA[r] = rintf(fminf(fmaxf(__expf(xv[r]-m)*linv, 0.f), 1.f) * 255.f);
    if (jb + 16 <= ib) {
      qk_tile(jb + 16, xv);
#pragma unroll
      for (int r = 0; r < 8; ++r)
        pB[r] = rintf(fminf(fmaxf(__expf(xv[r]-m)*linv, 0.f), 1.f) * 255.f);
    } else {
#pragma unroll
      for (int r = 0; r < 8; ++r) pB[r] = 0.f;
    }
    v16h Bf;
#pragma unroll
    for (int r = 0; r < 8; ++r) {
      float sA = __shfl_xor(pA[r], 16, 32);
      float sB = __shfl_xor(pB[r], 16, 32);
      float lo = half ? sB    : pA[r];
      float hi = half ? pB[r] : sA;
      Bf[r]   = (_Float16)lo;
      Bf[8+r] = (_Float16)hi;
    }
#pragma unroll
    for (int dt = 0; dt < 4; ++dt) {
      int d = dt*16 + ln;
      const _Float16* vp = vT + ((size_t)hkv*64 + d)*1024 + jb;
      v8i Ai;
#pragma unroll
      for (int r = 0; r < 8; ++r) Ai[r] = *(const int*)(vp + (half ? joff1[r] : joff0[r]));
      v16h Af = __builtin_bit_cast(v16h, Ai);
      oacc[dt] = __builtin_amdgcn_wmma_f32_16x16x32_f16(false, Af, false, Bf,
                                                        (short)0, oacc[dt], false, false);
    }
  }
  float* aorow = ao + (size_t)i*2048 + h*64;
#pragma unroll
  for (int dt = 0; dt < 4; ++dt)
#pragma unroll
    for (int r = 0; r < 8; ++r) {
      int d = dt*16 + r + half*8;
      aorow[d] = oacc[dt][r] * (1.0f/255.0f);
    }
}

extern "C" void kernel_launch(void* const* d_in, const int* in_sizes, int n_in,
                              void* d_out, int out_size, void* d_ws, size_t ws_size,
                              hipStream_t stream) {
  const float* hidden = (const float*)d_in[0];
  const float* cosb   = (const float*)d_in[1];
  const float* sinb   = (const float*)d_in[2];
  const int*   q_qw   = (const int*)  d_in[3];
  const float* q_sc   = (const float*)d_in[4];
  const int*   q_zp   = (const int*)  d_in[5];
  const int*   k_qw   = (const int*)  d_in[6];
  const float* k_sc   = (const float*)d_in[7];
  const int*   k_zp   = (const int*)  d_in[8];
  const int*   v_qw   = (const int*)  d_in[9];
  const float* v_sc   = (const float*)d_in[10];
  const int*   v_zp   = (const int*)  d_in[11];
  const int*   o_qw   = (const int*)  d_in[12];
  const float* o_sc   = (const float*)d_in[13];
  const int*   o_zp   = (const int*)  d_in[14];

  char* ws = (char*)d_ws;
  signed char* pw_qkv = (signed char*)(ws + OFF_PW_QKV);
  signed char* pw_o   = (signed char*)(ws + OFF_PW_O);
  float* scl_qkv  = (float*)(ws + OFF_SCL_QKV);
  float* scl_o    = (float*)(ws + OFF_SCL_O);
  float* wsum_qkv = (float*)(ws + OFF_WSUM_QKV);
  float* wsum_o   = (float*)(ws + OFF_WSUM_O);
  unsigned char* xq = (unsigned char*)(ws + OFF_XQ);
  float* xs = (float*)(ws + OFF_XS);
  float* xz = (float*)(ws + OFF_XZ);
  float* qkvf = (float*)(ws + OFF_QKVF);
  unsigned char* qq = (unsigned char*)(ws + OFF_QQ);
  float* qs  = (float*)(ws + OFF_QS);
  float* qz  = (float*)(ws + OFF_QZ);
  float* qsm = (float*)(ws + OFF_QSM);
  unsigned char* kq = (unsigned char*)(ws + OFF_KQ);
  float* ks  = (float*)(ws + OFF_KS);
  float* kz  = (float*)(ws + OFF_KZ);
  float* ksm = (float*)(ws + OFF_KSM);
  _Float16* vT = (_Float16*)(ws + OFF_VT);
  float* ao = (float*)(ws + OFF_AO);
  unsigned char* aq = (unsigned char*)(ws + OFF_AQ);
  float* as_ = (float*)(ws + OFF_AS);
  float* az  = (float*)(ws + OFF_AZ);

  pack_weights_kernel<<<5120, 128, 0, stream>>>(
      q_qw, q_sc, q_zp, k_qw, k_sc, k_zp, v_qw, v_sc, v_zp, o_qw, o_sc, o_zp,
      pw_qkv, scl_qkv, wsum_qkv, pw_o, scl_o, wsum_o);

  quant_act_kernel<<<1024, 256, 0, stream>>>(hidden, xq, xs, xz);

  gemm_u8s8_kernel<<<192, 256, 0, stream>>>(xq, xs, xz, pw_qkv, scl_qkv, wsum_qkv,
                                            qkvf, 1024, 3072);

  rope_quant_kernel<<<6144, 256, 0, stream>>>(qkvf, cosb, sinb,
                                              qq, qs, qz, qsm, kq, ks, kz, ksm, vT);

  attn_kernel<<<256, 256, 0, stream>>>(qq, qs, qz, qsm, kq, ks, kz, ksm, vT, ao);

  quant_act_kernel<<<1024, 256, 0, stream>>>(ao, aq, as_, az);

  gemm_u8s8_kernel<<<128, 256, 0, stream>>>(aq, as_, az, pw_o, scl_o, wsum_o,
                                            (float*)d_out, 1024, 2048);
}
